// myModel_4861902979248
// MI455X (gfx1250) — matrix operand order repaired and checked by reading, not compiled, not run on hardware
//
#include <hip/hip_runtime.h>
#include <hip/hip_bf16.h>
#include <math.h>

// ---------------------------------------------------------------------------
// Problem constants (match reference)
// ---------------------------------------------------------------------------
static constexpr int U_N   = 100000;
static constexpr int I_N   = 50000;
static constexpr int D_DIM = 64;
static constexpr int B_N   = 3;
static constexpr int NNZ_N = 500000;
static constexpr int L_N   = 2;

// LDS row padding (bf16 elements) for the transposed weight tile: row stride
// KTOT+8 -> 36 dwords for KTOT=64, making lane banks l*36 mod 64 all distinct
// (conflict-free ds_load_b128 across a 16-lane half).
static constexpr int WPAD = 8;

typedef __attribute__((ext_vector_type(16))) __bf16 v16bf;
typedef __attribute__((ext_vector_type(8)))  __bf16 v8bf;
typedef __attribute__((ext_vector_type(8)))  float  v8f;

// ---------------------------------------------------------------------------
// Utility kernels
// ---------------------------------------------------------------------------
__global__ void fill_zero_kernel(float* __restrict__ p, long long n4) {
  long long i = (long long)blockIdx.x * blockDim.x + threadIdx.x;
  if (i < n4) ((float4*)p)[i] = make_float4(0.f, 0.f, 0.f, 0.f);
}

__global__ void copy_kernel(const float* __restrict__ s, float* __restrict__ d,
                            long long n4) {
  long long i = (long long)blockIdx.x * blockDim.x + threadIdx.x;
  if (i < n4) ((float4*)d)[i] = ((const float4*)s)[i];
}

// ---------------------------------------------------------------------------
// SpMM (COO scatter):  Y[b, oidx[e], :] += vals[e] * X[iidx[e], :]
// 16 threads per edge, 4 floats each (float4 gather + 4 f32 atomics).
// grid.y = behavior b
// ---------------------------------------------------------------------------
__global__ void spmm_scatter_kernel(const int*  __restrict__ oidx,
                                    const int*  __restrict__ iidx,
                                    const float* __restrict__ vals,
                                    const float* __restrict__ X,
                                    float* __restrict__ Y,
                                    long long yBatchStride, int nnz) {
  const int b = blockIdx.y;
  const int*   ob = oidx + (long long)b * nnz;
  const int*   ib = iidx + (long long)b * nnz;
  const float* vb = vals + (long long)b * nnz;

  long long gid = (long long)blockIdx.x * blockDim.x + threadIdx.x;
  long long e = gid >> 4;
  if (e >= nnz) return;
  int j = ((int)gid & 15) * 4;

  int   r = ob[e];
  int   c = ib[e];
  float v = vb[e];
  float4 x = *(const float4*)(X + (long long)c * D_DIM + j);
  float* dst = Y + (long long)b * yBatchStride + (long long)r * D_DIM + j;
  atomicAdd(dst + 0, v * x.x);
  atomicAdd(dst + 1, v * x.y);
  atomicAdd(dst + 2, v * x.z);
  atomicAdd(dst + 3, v * x.w);
}

// ---------------------------------------------------------------------------
// WMMA fragment helpers (V_WMMA_F32_16X16X32_BF16, wave32)
//
// A (16x32 bf16) k order used here (contiguous, B matches): lanes 0-15 row M=lane, elems0-15 = K 0..15 ;
// lanes 16-31 row M=lane-16, elems0-15 = K 16..31.
// B (32x16 bf16) k order used here: lane = column; lanes 0-15 hold K 0..15 contiguously,
// lanes 16-31 hold K 16..31.
// C/D (16x16 f32): VGPR r -> row r (lanes 0-15) / row r+8 (lanes 16-31),
// lane&15 -> column.
// ---------------------------------------------------------------------------
__device__ __forceinline__ v16bf load_a_frag(const float* __restrict__ X,
                                             int ld, int row, int kbase) {
  const float* p = X + (long long)row * ld + kbase;
  v16bf a;
#pragma unroll
  for (int i = 0; i < 8; ++i) a[i] = (__bf16)p[i];
#pragma unroll
  for (int i = 0; i < 8; ++i) a[8 + i] = (__bf16)p[8 + i];
  return a;
}

// ldk = padded row stride (KTOT + WPAD); rows are 16-byte aligned, so load as
// two v8bf (ds_load_b128 each) and merge.
__device__ __forceinline__ v16bf load_b_frag(const __bf16* WT, int ldk,
                                             int col, int kbase) {
  const v8bf* p = (const v8bf*)(WT + (long long)col * ldk + kbase);
  v8bf lo = p[0];
  v8bf hi = p[1];
  return __builtin_shufflevector(lo, hi, 0, 1, 2, 3, 4, 5, 6, 7,
                                 8, 9, 10, 11, 12, 13, 14, 15);
}

__device__ __forceinline__ v8f wmma_bf16(v16bf a, v16bf b, v8f c) {
  return __builtin_amdgcn_wmma_f32_16x16x32_bf16(false, a, false, b,
                                                 (short)0, c, false, false);
}

// ---------------------------------------------------------------------------
// Fused GEMM: Y = [relu][l2norm-rows]( X[N,KTOT] @ W[KTOT,64] )
// 256 threads = 8 waves; wave computes 16 rows x 64 cols; block covers 128 rows.
// W staged transposed (bf16, padded rows) in LDS once per block.
// grid.y = batch index (with strides); wBatchStride may be 0.
// ---------------------------------------------------------------------------
template <int KTOT, bool RELU, bool L2N>
__global__ __launch_bounds__(256)
void wmma_gemm_kernel(const float* __restrict__ X, long long xBatchStride,
                      const float* __restrict__ W, long long wBatchStride,
                      float* __restrict__ Y0, long long y0BatchStride, int y0ld,
                      float* __restrict__ Y1, int y1ld,
                      int N) {
  static_assert(KTOT % 32 == 0, "KTOT must be a multiple of 32");
  constexpr int LDK = KTOT + WPAD;
  __shared__ __bf16 WT[64 * LDK];

  const int b = blockIdx.y;
  X  += (long long)b * xBatchStride;
  W  += (long long)b * wBatchStride;
  Y0 += (long long)b * y0BatchStride;

  for (int idx = threadIdx.x; idx < KTOT * 64; idx += blockDim.x) {
    int n = idx & 63, k = idx >> 6;
    WT[n * LDK + k] = (__bf16)W[(long long)k * 64 + n];
  }
  __syncthreads();

  const int wave = threadIdx.x >> 5;
  const int lane = threadIdx.x & 31;
  const int m0 = (blockIdx.x * 8 + wave) * 16;
  if (m0 >= N) return;                    // wave-uniform

  const int half = lane >> 4;
  int arow = m0 + (lane & 15);
  if (arow >= N) arow = N - 1;            // clamp (EXEC stays full for WMMA)

  v8f acc[4] = {};
#pragma unroll
  for (int kt = 0; kt < KTOT / 32; ++kt) {
    v16bf a = load_a_frag(X, KTOT, arow, kt * 32 + half * 16);
#pragma unroll
    for (int nt = 0; nt < 4; ++nt) {
      v16bf bm = load_b_frag(WT, LDK, nt * 16 + (lane & 15), kt * 32 + half * 16);
      acc[nt] = wmma_bf16(a, bm, acc[nt]);
    }
  }

  float inv[8];
  if (L2N) {
#pragma unroll
    for (int r = 0; r < 8; ++r) {
      float ss = 0.f;
#pragma unroll
      for (int nt = 0; nt < 4; ++nt) {
        float v = acc[nt][r];
        if (RELU) v = fmaxf(v, 0.f);
        ss += v * v;
      }
#pragma unroll
      for (int off = 1; off < 16; off <<= 1) ss += __shfl_xor(ss, off, 32);
      inv[r] = 1.0f / fmaxf(sqrtf(ss), 1e-12f);
    }
  }

  const int colb = lane & 15;
#pragma unroll
  for (int r = 0; r < 8; ++r) {
    int row = m0 + half * 8 + r;
    if (row >= N) continue;
#pragma unroll
    for (int nt = 0; nt < 4; ++nt) {
      float v = acc[nt][r];
      if (RELU) v = fmaxf(v, 0.f);
      if (L2N)  v *= inv[r];
      int col = nt * 16 + colb;
      Y0[(long long)row * y0ld + col] = v;
      if (Y1) Y1[(long long)row * y1ld + col] = v;
    }
  }
}

// ---------------------------------------------------------------------------
// Attention scores: S[b,n] = tanh(X[b,n,:] @ T1[b]) . T2[b]
// Same WMMA tiling; epilogue does in-register tanh, dot with T2, and a
// 16-lane butterfly reduction per row. grid.y = behavior b.
// ---------------------------------------------------------------------------
__global__ __launch_bounds__(256)
void wmma_scores_kernel(const float* __restrict__ X, long long xBatchStride,
                        const float* __restrict__ T1, long long t1BatchStride,
                        const float* __restrict__ T2, long long t2BatchStride,
                        float* __restrict__ S, long long sBatchStride,
                        int N) {
  constexpr int LDK = 64 + WPAD;
  __shared__ __bf16 WT[64 * LDK];
  __shared__ float t2s[64];

  const int b = blockIdx.y;
  X  += (long long)b * xBatchStride;
  T1 += (long long)b * t1BatchStride;
  T2 += (long long)b * t2BatchStride;
  S  += (long long)b * sBatchStride;

  for (int idx = threadIdx.x; idx < 64 * 64; idx += blockDim.x) {
    int n = idx & 63, k = idx >> 6;
    WT[n * LDK + k] = (__bf16)T1[(long long)k * 64 + n];
  }
  if (threadIdx.x < 64) t2s[threadIdx.x] = T2[threadIdx.x];
  __syncthreads();

  const int wave = threadIdx.x >> 5;
  const int lane = threadIdx.x & 31;
  const int m0 = (blockIdx.x * 8 + wave) * 16;
  if (m0 >= N) return;

  const int half = lane >> 4;
  int arow = m0 + (lane & 15);
  if (arow >= N) arow = N - 1;

  v8f acc[4] = {};
#pragma unroll
  for (int kt = 0; kt < 2; ++kt) {
    v16bf a = load_a_frag(X, 64, arow, kt * 32 + half * 16);
#pragma unroll
    for (int nt = 0; nt < 4; ++nt) {
      v16bf bm = load_b_frag(WT, LDK, nt * 16 + (lane & 15), kt * 32 + half * 16);
      acc[nt] = wmma_bf16(a, bm, acc[nt]);
    }
  }

  float t2l[4];
#pragma unroll
  for (int nt = 0; nt < 4; ++nt) t2l[nt] = t2s[nt * 16 + (lane & 15)];

  float myscore = 0.f;
#pragma unroll
  for (int r = 0; r < 8; ++r) {
    float p = 0.f;
#pragma unroll
    for (int nt = 0; nt < 4; ++nt) p += tanhf(acc[nt][r]) * t2l[nt];
#pragma unroll
    for (int off = 1; off < 16; off <<= 1) p += __shfl_xor(p, off, 32);
    if ((lane & 15) == r) myscore = p;
  }
  int row = m0 + half * 8 + (lane & 15);
  if ((lane & 15) < 8 && row < N) S[row] = myscore;
}

// ---------------------------------------------------------------------------
// Softmax over behaviors + combine:
//   w_b = softmax_b(S[:,n]) + 1/B ;  C[n,:] = sum_b w_b * AGG[b,n,:]
// 16 threads per row, float4 per thread.
// ---------------------------------------------------------------------------
__global__ void softmax_comb_kernel(const float* __restrict__ S,
                                    const float* __restrict__ AGG,
                                    float* __restrict__ C, int N) {
  long long gid = (long long)blockIdx.x * blockDim.x + threadIdx.x;
  long long n = gid >> 4;
  if (n >= N) return;
  int c = ((int)gid & 15) * 4;

  float s0 = S[n], s1 = S[(long long)N + n], s2 = S[2LL * N + n];
  float m = fmaxf(s0, fmaxf(s1, s2));
  float e0 = expf(s0 - m), e1 = expf(s1 - m), e2 = expf(s2 - m);
  float inv = 1.f / (e0 + e1 + e2);
  const float mb = 1.f / (float)B_N;
  float w0 = e0 * inv + mb, w1 = e1 * inv + mb, w2 = e2 * inv + mb;

  long long base = n * D_DIM + c;
  long long bs = (long long)N * D_DIM;
  float4 x0 = *(const float4*)(AGG + base);
  float4 x1 = *(const float4*)(AGG + bs + base);
  float4 x2 = *(const float4*)(AGG + 2 * bs + base);
  float4 o;
  o.x = w0 * x0.x + w1 * x1.x + w2 * x2.x;
  o.y = w0 * x0.y + w1 * x1.y + w2 * x2.y;
  o.z = w0 * x0.z + w1 * x1.z + w2 * x2.z;
  o.w = w0 * x0.w + w1 * x1.w + w2 * x2.w;
  *(float4*)(C + base) = o;
}

// ---------------------------------------------------------------------------
// Host-side orchestration
// ---------------------------------------------------------------------------
static inline int ceil_div_ll(long long a, long long b) { return (int)((a + b - 1) / b); }

extern "C" void kernel_launch(void* const* d_in, const int* in_sizes, int n_in,
                              void* d_out, int out_size, void* d_ws, size_t ws_size,
                              hipStream_t stream) {
  const float* user_table = (const float*)d_in[0];
  const float* item_table = (const float*)d_in[1];
  const int*   beh_rows   = (const int*)d_in[2];
  const int*   beh_cols   = (const int*)d_in[3];
  const float* beh_vals   = (const float*)d_in[4];
  const int*   comp_rows  = (const int*)d_in[5];
  const int*   comp_cols  = (const int*)d_in[6];
  const float* comp_vals  = (const float*)d_in[7];
  const float* u_w  = (const float*)d_in[8];
  const float* i_w  = (const float*)d_in[9];
  const float* s1   = (const float*)d_in[10];
  const float* s2   = (const float*)d_in[11];
  const float* s3   = (const float*)d_in[12];
  const float* s4   = (const float*)d_in[13];
  const float* u_cat = (const float*)d_in[14];
  const float* i_cat = (const float*)d_in[15];

  float* out = (float*)d_out;
  float* ws  = (float*)d_ws;

  // workspace layout (floats)
  long long off = 0;
  float* agg_u    = ws + off; off += (long long)B_N * U_N * D_DIM;
  float* agg_i    = ws + off; off += (long long)B_N * I_N * D_DIM;
  float* scores_u = ws + off; off += (long long)B_N * U_N;
  float* scores_i = ws + off; off += (long long)B_N * I_N;
  float* comb_u   = ws + off; off += (long long)U_N * D_DIM;
  float* comb_i   = ws + off; off += (long long)I_N * D_DIM;
  float* user_e   = ws + off; off += (long long)U_N * D_DIM;
  float* item_e   = ws + off; off += (long long)I_N * D_DIM;
  float* ue_cat   = ws + off; off += (long long)U_N * 2 * D_DIM;
  float* ie_cat   = ws + off; off += (long long)I_N * 2 * D_DIM;
  float* us_cat   = ws + off; off += (long long)B_N * U_N * 2 * D_DIM;
  float* is_cat   = ws + off; off += (long long)B_N * I_N * 2 * D_DIM;

  const int GU = ceil_div_ll(U_N, 128);   // row tiles of 128 (8 waves x 16)
  const int GI = ceil_div_ll(I_N, 128);

  // init embeddings
  copy_kernel<<<ceil_div_ll((long long)U_N * D_DIM / 4, 256), 256, 0, stream>>>(
      user_table, user_e, (long long)U_N * D_DIM / 4);
  copy_kernel<<<ceil_div_ll((long long)I_N * D_DIM / 4, 256), 256, 0, stream>>>(
      item_table, item_e, (long long)I_N * D_DIM / 4);

  for (int l = 0; l < L_N; ++l) {
    const int*   rows = (l == 0) ? beh_rows : comp_rows;
    const int*   cols = (l == 0) ? beh_cols : comp_cols;
    const float* vals = (l == 0) ? beh_vals : comp_vals;

    // zero aggregation buffers
    fill_zero_kernel<<<ceil_div_ll((long long)B_N * U_N * D_DIM / 4, 256), 256, 0, stream>>>(
        agg_u, (long long)B_N * U_N * D_DIM / 4);
    fill_zero_kernel<<<ceil_div_ll((long long)B_N * I_N * D_DIM / 4, 256), 256, 0, stream>>>(
        agg_i, (long long)B_N * I_N * D_DIM / 4);

    // SpMM both directions (reads previous user_e/item_e)
    {
      dim3 g(ceil_div_ll((long long)NNZ_N * 16, 256), B_N);
      spmm_scatter_kernel<<<g, 256, 0, stream>>>(rows, cols, vals, item_e, agg_u,
                                                 (long long)U_N * D_DIM, NNZ_N);
      spmm_scatter_kernel<<<g, 256, 0, stream>>>(cols, rows, vals, user_e, agg_i,
                                                 (long long)I_N * D_DIM, NNZ_N);
    }

    const float* uw_l = u_w + (long long)l * 64 * 64;
    const float* iw_l = i_w + (long long)l * 64 * 64;

    // ---- user side ----
    wmma_scores_kernel<<<dim3(GU, B_N), 256, 0, stream>>>(
        agg_u, (long long)U_N * D_DIM,
        s1 + (long long)l * B_N * 64 * 64, 64LL * 64,
        s2 + (long long)l * B_N * 64, 64LL,
        scores_u, (long long)U_N, U_N);
    softmax_comb_kernel<<<ceil_div_ll((long long)U_N * 16, 256), 256, 0, stream>>>(
        scores_u, agg_u, comb_u, U_N);
    // new user_e = l2norm(relu(comb @ uw)); also into ue_cat[:, l*64:]
    wmma_gemm_kernel<64, true, true><<<dim3(GU, 1), 256, 0, stream>>>(
        comb_u, 0, uw_l, 0, user_e, 0, 64, ue_cat + l * 64, 128, U_N);
    // us = relu(agg_u @ uw) into us_cat[b][:, l*64:]
    wmma_gemm_kernel<64, true, false><<<dim3(GU, B_N), 256, 0, stream>>>(
        agg_u, (long long)U_N * D_DIM, uw_l, 0,
        us_cat + l * 64, (long long)U_N * 128, 128, nullptr, 0, U_N);

    // ---- item side ----
    wmma_scores_kernel<<<dim3(GI, B_N), 256, 0, stream>>>(
        agg_i, (long long)I_N * D_DIM,
        s3 + (long long)l * B_N * 64 * 64, 64LL * 64,
        s4 + (long long)l * B_N * 64, 64LL,
        scores_i, (long long)I_N, I_N);
    softmax_comb_kernel<<<ceil_div_ll((long long)I_N * 16, 256), 256, 0, stream>>>(
        scores_i, agg_i, comb_i, I_N);
    wmma_gemm_kernel<64, true, true><<<dim3(GI, 1), 256, 0, stream>>>(
        comb_i, 0, iw_l, 0, item_e, 0, 64, ie_cat + l * 64, 128, I_N);
    wmma_gemm_kernel<64, true, false><<<dim3(GI, B_N), 256, 0, stream>>>(
        agg_i, (long long)I_N * D_DIM, iw_l, 0,
        is_cat + l * 64, (long long)I_N * 128, 128, nullptr, 0, I_N);
  }

  // ---- final concat projections ----
  float* out_ue  = out;                                            // [U,64]
  float* out_ie  = out + (long long)U_N * D_DIM;                   // [I,64]
  float* out_ues = out_ie + (long long)I_N * D_DIM;                // [B,U,64]
  float* out_ies = out_ues + (long long)B_N * U_N * D_DIM;         // [B,I,64]

  wmma_gemm_kernel<128, false, false><<<dim3(GU, 1), 256, 0, stream>>>(
      ue_cat, 0, u_cat, 0, out_ue, 0, 64, nullptr, 0, U_N);
  wmma_gemm_kernel<128, false, false><<<dim3(GI, 1), 256, 0, stream>>>(
      ie_cat, 0, i_cat, 0, out_ie, 0, 64, nullptr, 0, I_N);
  wmma_gemm_kernel<128, false, false><<<dim3(GU, B_N), 256, 0, stream>>>(
      us_cat, (long long)U_N * 128, u_cat, 0,
      out_ues, (long long)U_N * D_DIM, 64, nullptr, 0, U_N);
  wmma_gemm_kernel<128, false, false><<<dim3(GI, B_N), 256, 0, stream>>>(
      is_cat, (long long)I_N * 128, i_cat, 0,
      out_ies, (long long)I_N * D_DIM, 64, nullptr, 0, I_N);
}
